// Decoder_87505663689049
// MI455X (gfx1250) — matrix operand order repaired and checked by reading, not compiled, not run on hardware
//
#include <hip/hip_runtime.h>

#define TDEC   500
#define TENC   512
#define BATCH  64
#define EDIM   512
#define ADIM   128
#define HDIM   1024
#define PREDIM 256
#define NMEL   80

typedef __attribute__((ext_vector_type(16))) __bf16 v16bf;
typedef __attribute__((ext_vector_type(8)))  __bf16 v8bf;
typedef __attribute__((ext_vector_type(8)))  float  v8f;

// ---- WMMA helpers (CDNA5 16x16x32 bf16, wave32) -----------------------------
// A tile 16x32 (MxK): lane m (0-15) rows; elems 0..15 = K 16h..16h+15
__device__ __forceinline__ v16bf loadA16x32(const __bf16* base, int ldk) {
  int lane = threadIdx.x & 31;
  int hh = lane >> 4, m = lane & 15;
  const __bf16* p = base + (size_t)m * ldk + 16 * hh;
  union { v16bf v; v8bf h[2]; } r;
  r.h[0] = *(const v8bf*)p;
  r.h[1] = *(const v8bf*)(p + 8);
  return r.v;
}
// B tile 32x16 (KxN): lane n (0-15) = column n; elems 0..15 = K 16h..16h+15.
// B[k][n] = W[n][k] (row-major weight, contiguous along k).
__device__ __forceinline__ v16bf loadB32x16(const __bf16* base, int ldk) {
  int lane = threadIdx.x & 31;
  int hh = lane >> 4, n = lane & 15;
  return *(const v16bf*)(base + (size_t)n * ldk + 16 * hh);
}
__device__ __forceinline__ v8f wmma_bf16(v16bf a, v16bf b, v8f c) {
  // D = A x B + C, f32 accum
  return __builtin_amdgcn_wmma_f32_16x16x32_bf16(false, a, false, b, (short)0, c, false, false);
}

// ---- generic GEMM: out[M,N] = act(A[M,K](bf16) @ W[N,K]^T(bf16)) ------------
// grid = (M/64, N/64), block = 256 (8 waves: 4 col blocks x 2 row pairs)
__global__ __launch_bounds__(256) void gemm_bf16(
    const __bf16* __restrict__ A, int lda,
    const __bf16* __restrict__ W, int ldw,
    __bf16* outBf, float* outF,
    int N, int K, int relu)
{
  int m0 = blockIdx.x * 64;
  int n0 = blockIdx.y * 64;
  int wave = threadIdx.x >> 5;
  int cb = wave & 3, rp = wave >> 2;
  int nn0 = n0 + cb * 16;
  v8f z = {0.f,0.f,0.f,0.f,0.f,0.f,0.f,0.f};
  v8f acc0 = z, acc1 = z;
  const __bf16* a0base = A + (size_t)(m0 + rp * 32) * lda;
  const __bf16* a1base = A + (size_t)(m0 + rp * 32 + 16) * lda;
  const __bf16* wbase  = W + (size_t)nn0 * ldw;
  for (int k0 = 0; k0 < K; k0 += 32) {
    v16bf a0 = loadA16x32(a0base + k0, lda);
    v16bf a1 = loadA16x32(a1base + k0, lda);
    v16bf b  = loadB32x16(wbase + k0, ldw);
    acc0 = wmma_bf16(a0, b, acc0);
    acc1 = wmma_bf16(a1, b, acc1);
  }
  int lane = threadIdx.x & 31;
  int hh = lane >> 4, nl = lane & 15;
  int col = nn0 + nl;
  for (int r = 0; r < 8; r++) {
    int row0 = m0 + rp * 32 + r + 8 * hh;
    int row1 = row0 + 16;
    float v0 = acc0[r], v1 = acc1[r];
    if (relu) { v0 = fmaxf(v0, 0.f); v1 = fmaxf(v1, 0.f); }
    if (outBf) {
      outBf[(size_t)row0 * N + col] = (__bf16)v0;
      outBf[(size_t)row1 * N + col] = (__bf16)v1;
    }
    if (outF) {
      outF[(size_t)row0 * N + col] = v0;
      outF[(size_t)row1 * N + col] = v1;
    }
  }
}

// ---- fused LSTM step: gates[64,4096] = x@Wih^T + h@Whh^T; apply cell --------
// x is a segmented concat of up to 3 bf16 sources (segment lens multiple of 32).
// grid = 64 (16 gate-columns each), block = 128 (4 waves = 4 row blocks of 16).
__global__ __launch_bounds__(128) void lstm_step(
    const __bf16* __restrict__ x0, int len0, int ld0,
    const __bf16* __restrict__ x1, int len1, int ld1,
    const __bf16* __restrict__ x2, int len2, int ld2,
    const __bf16* __restrict__ wih, int Kih,
    const __bf16* __restrict__ whh,
    const float* __restrict__ bih, const float* __restrict__ bhh,
    float* __restrict__ hst, float* __restrict__ cst,
    __bf16* __restrict__ hbf)
{
  const int H = HDIM;
  int c0 = blockIdx.x * 16;
  int rp = threadIdx.x >> 5;
  int K = len0 + len1 + len2;
  v8f z = {0.f,0.f,0.f,0.f,0.f,0.f,0.f,0.f};
  v8f acc[4];
  for (int g = 0; g < 4; g++) acc[g] = z;
  for (int k0 = 0; k0 < K; k0 += 32) {
    const __bf16* pa; int lda, kk;
    if (k0 < len0)             { pa = x0; lda = ld0; kk = k0; }
    else if (k0 < len0 + len1) { pa = x1; lda = ld1; kk = k0 - len0; }
    else                       { pa = x2; lda = ld2; kk = k0 - len0 - len1; }
    v16bf a = loadA16x32(pa + (size_t)(rp * 16) * lda + kk, lda);
    const __bf16* pb; int ldb, kb;
    if (k0 < Kih) { pb = wih; ldb = Kih; kb = k0; }
    else          { pb = whh; ldb = H;   kb = k0 - Kih; }
    for (int g = 0; g < 4; g++) {
      v16bf b = loadB32x16(pb + (size_t)(g * H + c0) * ldb + kb, ldb);
      acc[g] = wmma_bf16(a, b, acc[g]);
    }
  }
  int lane = threadIdx.x & 31;
  int hh = lane >> 4, nl = lane & 15;
  int col = c0 + nl;
  float b0 = bih[col]         + bhh[col];
  float b1 = bih[H + col]     + bhh[H + col];
  float b2 = bih[2 * H + col] + bhh[2 * H + col];
  float b3 = bih[3 * H + col] + bhh[3 * H + col];
  for (int r = 0; r < 8; r++) {
    int row = rp * 16 + r + 8 * hh;
    size_t idx = (size_t)row * H + col;
    float iv = 1.f / (1.f + __expf(-(acc[0][r] + b0)));
    float fv = 1.f / (1.f + __expf(-(acc[1][r] + b1)));
    float gv = tanhf(acc[2][r] + b2);
    float ov = 1.f / (1.f + __expf(-(acc[3][r] + b3)));
    float cn = fv * cst[idx] + iv * gv;
    float hn = ov * tanhf(cn);
    cst[idx] = cn;
    hst[idx] = hn;
    hbf[idx] = (__bf16)hn;
  }
}

// ---- attention step: conv -> (WMMA loc) -> energies -> softmax -> context ---
// grid = 64 (one block per batch row), block = 512 (16 waves)
__global__ __launch_bounds__(512) void attn_step(
    const float* __restrict__ ah,        // [64,1024]
    const float* __restrict__ wq,        // [128,1024]
    const float* __restrict__ attn_v,    // [128]
    const float* __restrict__ loc_conv,  // [32,2,31]
    const __bf16* __restrict__ locwbf,   // [128,32]
    const float* __restrict__ pm,        // [64,512,128]
    const float* __restrict__ memory,    // [64,512,512]
    const int*   __restrict__ mlen,      // [64]
    float* __restrict__ aw_g, float* __restrict__ awc_g,
    float* __restrict__ ctx_g, __bf16* __restrict__ ctxbf,
    float* __restrict__ align_out, int step)
{
  __shared__ float aw_s[TENC], awc_s[TENC], e_s[TENC];
  __shared__ float q_s[ADIM], v_s[ADIM];
  __shared__ float lc_s[32 * 2 * 31];
  __shared__ __bf16 cbf[TENC * 32];
  __shared__ float red_s[32];

  int b = blockIdx.x;
  int tid = threadIdx.x;
  int wave = tid >> 5, lane = tid & 31;
  int hh = lane >> 4, nl = lane & 15;

  aw_s[tid]  = aw_g[(size_t)b * TENC + tid];
  awc_s[tid] = awc_g[(size_t)b * TENC + tid];
  e_s[tid]   = 0.f;
  for (int i = tid; i < 32 * 2 * 31; i += 512) lc_s[i] = loc_conv[i];
  if (tid < ADIM) {
    v_s[tid] = attn_v[tid];
    float q = 0.f;
    const float* ar = ah + (size_t)b * HDIM;
    const float* wr = wq + (size_t)tid * HDIM;
    for (int k = 0; k < HDIM; k++) q += ar[k] * wr[k];
    q_s[tid] = q;
  }
  __syncthreads();

  // 31-tap conv over [aw; awc], 32 filters; result -> bf16 in LDS
  {
    int t = tid;
    float c[32];
    for (int f = 0; f < 32; f++) c[f] = 0.f;
    for (int k = 0; k < 31; k++) {
      int idx = t + k - 15;
      float a = 0.f, w = 0.f;
      if (idx >= 0 && idx < TENC) { a = aw_s[idx]; w = awc_s[idx]; }
      for (int f = 0; f < 32; f++)
        c[f] += lc_s[f * 62 + k] * a + lc_s[f * 62 + 31 + k] * w;
    }
    for (int f = 0; f < 32; f++) cbf[t * 32 + f] = (__bf16)c[f];
  }
  __syncthreads();

  // loc = conv @ loc_w^T via WMMA (K=32), fused tanh-energy reduction
  const float* pmb = pm + (size_t)b * TENC * ADIM;
  for (int st = wave; st < (TENC / 16) * (ADIM / 16); st += 16) {
    int rt = st >> 3, ct = st & 7;
    int t0 = rt * 16, n0 = ct * 16;
    v16bf a  = loadA16x32(&cbf[t0 * 32], 32);
    v16bf bb = loadB32x16(locwbf + (size_t)n0 * 32, 32);
    v8f d = {0.f,0.f,0.f,0.f,0.f,0.f,0.f,0.f};
    d = wmma_bf16(a, bb, d);
    int nn = n0 + nl;
    for (int r = 0; r < 8; r++) {
      int tt = t0 + r + 8 * hh;
      float e = tanhf(q_s[nn] + d[r] + pmb[(size_t)tt * ADIM + nn]) * v_s[nn];
      for (int o = 8; o; o >>= 1) e += __shfl_xor(e, o, 16);
      if (nl == 0) atomicAdd(&e_s[tt], e);
    }
  }
  __syncthreads();

  // masked softmax over T_ENC
  int L = mlen[b];
  float ev = (tid < L) ? e_s[tid] : -3.402823466e38f;
  float m = ev;
  for (int o = 16; o; o >>= 1) m = fmaxf(m, __shfl_xor(m, o, 32));
  if (lane == 0) red_s[wave] = m;
  __syncthreads();
  if (tid == 0) { float x = red_s[0]; for (int i = 1; i < 16; i++) x = fmaxf(x, red_s[i]); red_s[0] = x; }
  __syncthreads();
  float mx = red_s[0];
  float p = (tid < L) ? __expf(ev - mx) : 0.f;
  __syncthreads();
  float sw = p;
  for (int o = 16; o; o >>= 1) sw += __shfl_xor(sw, o, 32);
  if (lane == 0) red_s[wave] = sw;
  __syncthreads();
  if (tid == 0) { float x = 0.f; for (int i = 0; i < 16; i++) x += red_s[i]; red_s[0] = x; }
  __syncthreads();
  float awn = p / red_s[0];

  aw_s[tid] = awn;
  aw_g[(size_t)b * TENC + tid] = awn;
  awc_g[(size_t)b * TENC + tid] = awc_s[tid] + awn;
  align_out[((size_t)b * TDEC + step) * TENC + tid] = awn;
  __syncthreads();

  // ctx = aw @ memory  (f32, coalesced over E)
  {
    int e = tid;
    float s = 0.f;
    const float* mb = memory + (size_t)b * TENC * EDIM;
    for (int t = 0; t < TENC; t++) s += aw_s[t] * mb[(size_t)t * EDIM + e];
    ctx_g[(size_t)b * EDIM + e] = s;
    ctxbf[(size_t)b * EDIM + e] = (__bf16)s;
  }
}

// ---- projection / gate ------------------------------------------------------
__global__ __launch_bounds__(128) void proj_step(
    const float* __restrict__ dh, const float* __restrict__ ctx,
    const float* __restrict__ proj_w, const float* __restrict__ proj_b,
    const float* __restrict__ gate_w, const float* __restrict__ gate_b,
    float* __restrict__ mel_out, float* __restrict__ gate_out, int step)
{
  int b = blockIdx.x, n = threadIdx.x;
  if (n > NMEL) return;
  const float* w = (n < NMEL) ? (proj_w + (size_t)n * (HDIM + EDIM)) : gate_w;
  float acc = (n < NMEL) ? proj_b[n] : gate_b[0];
  const float* dhb = dh + (size_t)b * HDIM;
  const float* cb  = ctx + (size_t)b * EDIM;
  for (int k = 0; k < HDIM; k++) acc += w[k] * dhb[k];
  for (int k = 0; k < EDIM; k++) acc += w[HDIM + k] * cb[k];
  if (n < NMEL) mel_out[((size_t)b * NMEL + n) * TDEC + step] = acc;
  else          gate_out[(size_t)b * TDEC + step] = acc;
}

// ---- setup / conversion kernels --------------------------------------------
__global__ void f2bf(const float* __restrict__ s, __bf16* __restrict__ d, size_t n) {
  size_t stride = (size_t)gridDim.x * blockDim.x;
  for (size_t i = (size_t)blockIdx.x * blockDim.x + threadIdx.x; i < n; i += stride)
    d[i] = (__bf16)s[i];
}
__global__ void padw1_k(const float* __restrict__ src, __bf16* __restrict__ dst) {
  int i = blockIdx.x * blockDim.x + threadIdx.x;
  if (i >= 256 * 96) return;
  int r = i / 96, k = i % 96;
  dst[i] = (k < NMEL) ? (__bf16)src[r * NMEL + k] : (__bf16)0.f;
}
// decbf[t][b][k96] = (t==0 || k>=80) ? 0 : decoder_inputs[b][k][t-1]
__global__ void decbf_k(const float* __restrict__ dec, __bf16* __restrict__ dst) {
  size_t n = (size_t)TDEC * BATCH * 96;
  size_t stride = (size_t)gridDim.x * blockDim.x;
  for (size_t i = (size_t)blockIdx.x * blockDim.x + threadIdx.x; i < n; i += stride) {
    int k = (int)(i % 96);
    size_t r = i / 96;
    int b = (int)(r % BATCH);
    int t = (int)(r / BATCH);
    float v = 0.f;
    if (t > 0 && k < NMEL) v = dec[((size_t)b * NMEL + k) * TDEC + (t - 1)];
    dst[i] = (__bf16)v;
  }
}

// ---- host driver ------------------------------------------------------------
extern "C" void kernel_launch(void* const* d_in, const int* in_sizes, int n_in,
                              void* d_out, int out_size, void* d_ws, size_t ws_size,
                              hipStream_t stream)
{
  (void)in_sizes; (void)n_in; (void)out_size; (void)ws_size;
  const float* memory    = (const float*)d_in[0];
  const float* dec       = (const float*)d_in[1];
  const int*   mlen      = (const int*)  d_in[2];
  const float* pre_w1    = (const float*)d_in[3];
  const float* pre_w2    = (const float*)d_in[4];
  const float* arnn_wih  = (const float*)d_in[5];
  const float* arnn_whh  = (const float*)d_in[6];
  const float* arnn_bih  = (const float*)d_in[7];
  const float* arnn_bhh  = (const float*)d_in[8];
  const float* wq        = (const float*)d_in[9];
  const float* wmem      = (const float*)d_in[10];
  const float* vvec      = (const float*)d_in[11];
  const float* loc_conv  = (const float*)d_in[12];
  const float* loc_w     = (const float*)d_in[13];
  const float* drnn_wih  = (const float*)d_in[14];
  const float* drnn_whh  = (const float*)d_in[15];
  const float* drnn_bih  = (const float*)d_in[16];
  const float* drnn_bhh  = (const float*)d_in[17];
  const float* proj_w    = (const float*)d_in[18];
  const float* proj_b    = (const float*)d_in[19];
  const float* gate_w    = (const float*)d_in[20];
  const float* gate_b    = (const float*)d_in[21];

  char* ws = (char*)d_ws;
  size_t off = 0;
  auto alc = [&](size_t bytes) -> char* {
    char* p = ws + off;
    off += (bytes + 255) & ~(size_t)255;
    return p;
  };

  __bf16* membf  = (__bf16*)alc((size_t)BATCH * TENC * EDIM * 2);
  float*  pmf    = (float*) alc((size_t)BATCH * TENC * ADIM * 4);
  __bf16* decbf  = (__bf16*)alc((size_t)TDEC * BATCH * 96 * 2);
  __bf16* h1bf   = (__bf16*)alc((size_t)TDEC * BATCH * PREDIM * 2);
  __bf16* prebf  = (__bf16*)alc((size_t)TDEC * BATCH * PREDIM * 2);
  __bf16* w1bf   = (__bf16*)alc((size_t)256 * 96 * 2);
  __bf16* w2bf   = (__bf16*)alc((size_t)256 * 256 * 2);
  __bf16* wmembf = (__bf16*)alc((size_t)ADIM * EDIM * 2);
  __bf16* awihbf = (__bf16*)alc((size_t)4096 * 768 * 2);
  __bf16* awhhbf = (__bf16*)alc((size_t)4096 * 1024 * 2);
  __bf16* dwihbf = (__bf16*)alc((size_t)4096 * 1536 * 2);
  __bf16* dwhhbf = (__bf16*)alc((size_t)4096 * 1024 * 2);
  __bf16* locwbf = (__bf16*)alc((size_t)ADIM * 32 * 2);

  char* state_base = ws + off;
  float* ah  = (float*)alc((size_t)BATCH * HDIM * 4);
  float* ac  = (float*)alc((size_t)BATCH * HDIM * 4);
  float* dhs = (float*)alc((size_t)BATCH * HDIM * 4);
  float* dcs = (float*)alc((size_t)BATCH * HDIM * 4);
  float* aw  = (float*)alc((size_t)BATCH * TENC * 4);
  float* awc = (float*)alc((size_t)BATCH * TENC * 4);
  float* ctx = (float*)alc((size_t)BATCH * EDIM * 4);
  __bf16* ahbf0 = (__bf16*)alc((size_t)BATCH * HDIM * 2);
  __bf16* ahbf1 = (__bf16*)alc((size_t)BATCH * HDIM * 2);
  __bf16* dhbf0 = (__bf16*)alc((size_t)BATCH * HDIM * 2);
  __bf16* dhbf1 = (__bf16*)alc((size_t)BATCH * HDIM * 2);
  __bf16* ctxbf = (__bf16*)alc((size_t)BATCH * EDIM * 2);
  size_t state_bytes = (size_t)((ws + off) - state_base);

  hipMemsetAsync(state_base, 0, state_bytes, stream);

  auto cvt = [&](const float* s, __bf16* d, size_t n) {
    int blocks = (int)((n + 255) / 256);
    if (blocks > 8192) blocks = 8192;
    f2bf<<<blocks, 256, 0, stream>>>(s, d, n);
  };
  cvt(memory,   membf,  (size_t)BATCH * TENC * EDIM);
  cvt(pre_w2,   w2bf,   (size_t)256 * 256);
  cvt(wmem,     wmembf, (size_t)ADIM * EDIM);
  cvt(arnn_wih, awihbf, (size_t)4096 * 768);
  cvt(arnn_whh, awhhbf, (size_t)4096 * 1024);
  cvt(drnn_wih, dwihbf, (size_t)4096 * 1536);
  cvt(drnn_whh, dwhhbf, (size_t)4096 * 1024);
  cvt(loc_w,    locwbf, (size_t)ADIM * 32);
  padw1_k<<<(256 * 96 + 255) / 256, 256, 0, stream>>>(pre_w1, w1bf);
  decbf_k<<<8192, 256, 0, stream>>>(dec, decbf);

  // prenet (all timesteps, parallel) and processed memory, via WMMA
  gemm_bf16<<<dim3(TDEC, 4), 256, 0, stream>>>(decbf, 96,  w1bf, 96,  h1bf,  nullptr, PREDIM, 96,  1);
  gemm_bf16<<<dim3(TDEC, 4), 256, 0, stream>>>(h1bf,  256, w2bf, 256, prebf, nullptr, PREDIM, 256, 1);
  gemm_bf16<<<dim3(512, 2),  256, 0, stream>>>(membf, 512, wmembf, 512, nullptr, pmf,  ADIM,  512, 0);

  float* out = (float*)d_out;
  float* mel_out   = out;
  float* gate_out  = out + (size_t)BATCH * NMEL * TDEC;
  float* align_out = gate_out + (size_t)BATCH * TDEC;

  for (int t = 0; t < TDEC; t++) {
    const __bf16* ain = (t & 1) ? ahbf1 : ahbf0;
    __bf16*       aou = (t & 1) ? ahbf0 : ahbf1;
    const __bf16* din = (t & 1) ? dhbf1 : dhbf0;
    __bf16*       dou = (t & 1) ? dhbf0 : dhbf1;

    // attention LSTM: x = [pre_t(256) | ctx(512)], h = ah(1024)
    lstm_step<<<64, 128, 0, stream>>>(
        prebf + (size_t)t * BATCH * PREDIM, PREDIM, PREDIM,
        ctxbf, EDIM, EDIM,
        ain, HDIM, HDIM,
        awihbf, 768, awhhbf,
        arnn_bih, arnn_bhh, ah, ac, aou);

    attn_step<<<BATCH, 512, 0, stream>>>(
        ah, wq, vvec, loc_conv, locwbf, pmf, memory, mlen,
        aw, awc, ctx, ctxbf, align_out, t);

    // decoder LSTM: x = [ah(1024) | ctx(512)], h = dh(1024)
    lstm_step<<<64, 128, 0, stream>>>(
        aou, HDIM, HDIM,
        ctxbf, EDIM, EDIM,
        din, HDIM, HDIM,
        dwihbf, 1536, dwhhbf,
        drnn_bih, drnn_bhh, dhs, dcs, dou);

    proj_step<<<BATCH, 128, 0, stream>>>(
        dhs, ctx, proj_w, proj_b, gate_w, gate_b, mel_out, gate_out, t);
  }
}
